// MetaMasker_91036126806754
// MI455X (gfx1250) — matrix operand order repaired and checked by reading, not compiled, not run on hardware
//
#include <hip/hip_runtime.h>
#include <hip/hip_bf16.h>

typedef __bf16 v16bf __attribute__((ext_vector_type(16)));
typedef __bf16 bf8   __attribute__((ext_vector_type(8)));
typedef float  v8f   __attribute__((ext_vector_type(8)));

#define B_   32
#define T_   64
#define N_   256
#define F_   64
#define NTF  (T_*N_*F_)          // 2^20 per batch
#define TOTAL ((size_t)B_*NTF)   // 33554432
#define KTOP 209715
#define NBIN 4096

union V16U { bf8 h[2]; v16bf v; };

__device__ __forceinline__ v8f zero8() {
    v8f z = {0.f,0.f,0.f,0.f,0.f,0.f,0.f,0.f};
    return z;
}

// A tile (16x32 bf16, MxK): lane<16 row M=lane, elems {K0..7, K16..23};
// lane>=16 same rows, elems {K8..15, K24..31}
__device__ __forceinline__ v16bf load_a16(const __bf16* base, int stride, int m0, int k0, int lane) {
    int h = lane >> 4, r = lane & 15;
    const __bf16* row = base + (size_t)(m0 + r) * stride + k0 + h * 8;
    V16U u;
    u.h[0] = *(const bf8*)(row);
    u.h[1] = *(const bf8*)(row + 16);
    return u.v;
}

// B tile (32x16 bf16, KxN) from a row-major "B-transposed" array (row n holds K
// contiguous): lane<16 col N=lane, K={k0..k0+7, k0+16..k0+23}; lane>=16 same cols, K={k0+8..k0+15, k0+24..k0+31}
__device__ __forceinline__ v16bf load_bt16(const __bf16* bt, int stride, int n0, int k0, int lane) {
    int h = lane >> 4, r = lane & 15;
    const __bf16* row = bt + (size_t)(n0 + r) * stride + k0 + h * 8;
    V16U u;
    u.h[0] = *(const bf8*)(row);
    u.h[1] = *(const bf8*)(row + 16);
    return u.v;
}

__device__ __forceinline__ v8f wmma_bf16(v16bf a, v16bf b, v8f c) {
    return __builtin_amdgcn_wmma_f32_16x16x32_bf16(false, a, false, b, (short)0, c, false, false);
}

// ---------------- kernel 0: zero histograms ----------------
__global__ void mm_zero(int* p, int n) {
    int i = blockIdx.x * blockDim.x + threadIdx.x;
    if (i < n) p[i] = 0;
}

// ---------------- kernel 1: weight prep (bf16, transposed; conv shifts merged) --
__global__ void mm_prep(const float* Wc1, const float* Wc2, const float* Wc3,
                        const float* bc1, const float* bc2, const float* bc3,
                        const float* Wq, const float* Wk, const float* Wv,
                        const float* Wd, const float* Wp,
                        __bf16* wt, float* bconv) {
    int i = blockIdx.x * blockDim.x + threadIdx.x;
    if (i < 12 * 4096) {
        int w = i >> 12, e = i & 4095;
        int n = e >> 6, k = e & 63;
        int src = k * 64 + n;                 // transpose: wt[n][k] = W[k][n]
        float v = 0.f;
        switch (w) {
            case 0:  v = Wc3[src];            break;   // shift -3
            case 1:  v = Wc2[src];            break;   // shift -2
            case 2:  v = Wc1[src];            break;   // shift -1
            case 3:  v = Wc1[4096 + src] + Wc2[4096 + src] + Wc3[4096 + src]; break; // shift 0
            case 4:  v = Wc1[2*4096 + src];   break;   // shift +1
            case 5:  v = Wc2[2*4096 + src];   break;   // shift +2
            case 6:  v = Wc3[2*4096 + src];   break;   // shift +3
            case 7:  v = Wq[src]; break;
            case 8:  v = Wk[src]; break;
            case 9:  v = Wv[src]; break;
            case 10: v = Wd[src]; break;
            case 11: v = Wp[src]; break;
        }
        wt[i] = (__bf16)v;
    } else if (i < 12 * 4096 + 64) {
        int f = i - 12 * 4096;
        bconv[f] = bc1[f] + bc2[f] + bc3[f];
    }
}

// ---------------- kernel 2: dilated conv block as 7 shifted GEMMs + residual ----
__global__ __launch_bounds__(128) void mm_conv(const float* __restrict__ x,
                                               const __bf16* __restrict__ wt,
                                               const float* __restrict__ bconv,
                                               __bf16* __restrict__ x1) {
    __shared__ __align__(16) __bf16 xs[70 * 64];   // time-padded (+/-3) bf16 slab
    int bid = blockIdx.x;            // b*N + n
    int b = bid >> 8, n = bid & 255;
    int tid = threadIdx.x;
    const float* xbase = x + ((size_t)b * T_) * N_ * F_ + (size_t)n * F_;
    for (int e = tid; e < 70 * 64; e += 128) {
        int row = e >> 6, f = e & 63;
        int t = row - 3;
        float v = (t >= 0 && t < T_) ? xbase[(size_t)t * N_ * F_ + f] : 0.f;
        xs[e] = (__bf16)v;
    }
    __syncthreads();

    int wid = tid >> 5, lane = tid & 31;
    int h = lane >> 4, r15 = lane & 15;
    int m0 = wid * 16;               // tile-row over T
    v8f acc[4];
#pragma unroll
    for (int j = 0; j < 4; ++j) acc[j] = zero8();

    for (int s = 0; s < 7; ++s) {    // effective time shift s-3
        const __bf16* W = wt + s * 4096;
#pragma unroll
        for (int kk = 0; kk < 2; ++kk) {
            int k0 = kk * 32;
            v16bf a = load_a16(xs, 64, m0 + s, k0, lane);
#pragma unroll
            for (int j = 0; j < 4; ++j)
                acc[j] = wmma_bf16(a, load_bt16(W, 64, j * 16, k0, lane), acc[j]);
        }
    }
#pragma unroll
    for (int j = 0; j < 4; ++j) {
        int f = j * 16 + r15;
        float bias = bconv[f];
#pragma unroll
        for (int r = 0; r < 8; ++r) {
            int t = m0 + h * 8 + r;
            float conv = acc[j][r] + bias;
            float out = xbase[(size_t)t * N_ * F_ + f] + fmaxf(conv, 0.f);
            x1[(((size_t)b * T_ + t) * N_ + n) * F_ + f] = (__bf16)out;
        }
    }
}

// ---------------- kernel 3: QKV + diag-softmax GAT + Wd + inf-norm + hist ------
__global__ __launch_bounds__(256) void mm_attn(const __bf16* __restrict__ x1,
                                               const __bf16* __restrict__ wt,
                                               const float* __restrict__ bq,
                                               const float* __restrict__ bk,
                                               const float* __restrict__ bv,
                                               const float* __restrict__ bd,
                                               float* __restrict__ x4,
                                               int* __restrict__ hist1) {
    __shared__ __align__(16) __bf16 qs[256 * 64];  // Q, later x2
    __shared__ __align__(16) __bf16 ks[256 * 64];  // K, later diagA (first 1KB)
    int bt = blockIdx.x;
    int b  = bt >> 6;                 // T_=64
    const __bf16* xs = x1 + (size_t)bt * (N_ * F_);
    int tid = threadIdx.x, wid = tid >> 5, lane = tid & 31;
    int h = lane >> 4, r15 = lane & 15;
    const __bf16* Wq = wt + 7 * 4096;
    const __bf16* Wk = wt + 8 * 4096;
    const __bf16* Wv = wt + 9 * 4096;
    const __bf16* Wd = wt + 10 * 4096;

    // ---- Q, K projections (A tiles straight from global X1 slab) ----
    for (int trr = 0; trr < 2; ++trr) {
        int m0 = wid * 32 + trr * 16;
        v8f aq[4], ak[4];
#pragma unroll
        for (int j = 0; j < 4; ++j) { aq[j] = zero8(); ak[j] = zero8(); }
#pragma unroll
        for (int kk = 0; kk < 2; ++kk) {
            int k0 = kk * 32;
            v16bf a = load_a16(xs, 64, m0, k0, lane);
#pragma unroll
            for (int j = 0; j < 4; ++j) {
                aq[j] = wmma_bf16(a, load_bt16(Wq, 64, j * 16, k0, lane), aq[j]);
                ak[j] = wmma_bf16(a, load_bt16(Wk, 64, j * 16, k0, lane), ak[j]);
            }
        }
#pragma unroll
        for (int j = 0; j < 4; ++j) {
            int f = j * 16 + r15;
            float bqv = bq[f], bkv = bk[f];
#pragma unroll
            for (int r = 0; r < 8; ++r) {
                int nr = m0 + h * 8 + r;
                qs[nr * 64 + f] = (__bf16)(aq[j][r] + bqv);
                ks[nr * 64 + f] = (__bf16)(ak[j][r] + bkv);
            }
        }
    }
    __syncthreads();

    // ---- scores Q*K^T with online softmax; keep only diag(A) ----
    float dA[2][8];
    for (int trr = 0; trr < 2; ++trr) {
        int m0 = wid * 32 + trr * 16;
        int tr = m0 >> 4;
        v16bf a0 = load_a16(qs, 64, m0, 0, lane);
        v16bf a1 = load_a16(qs, 64, m0, 32, lane);
        float mr[8], sr[8], dg[8];
#pragma unroll
        for (int r = 0; r < 8; ++r) { mr[r] = -1e30f; sr[r] = 0.f; dg[r] = 0.f; }
        for (int tc = 0; tc < 16; ++tc) {
            v8f sc = zero8();
            sc = wmma_bf16(a0, load_bt16(ks, 64, tc * 16, 0,  lane), sc);
            sc = wmma_bf16(a1, load_bt16(ks, 64, tc * 16, 32, lane), sc);
#pragma unroll
            for (int r = 0; r < 8; ++r) {
                float v = sc[r] * 0.125f;          // 1/sqrt(64)
                if (tc == tr && r15 == h * 8 + r) dg[r] = v;
                float mn = fmaxf(mr[r], v);
                sr[r] = sr[r] * __expf(mr[r] - mn) + __expf(v - mn);
                mr[r] = mn;
            }
        }
#pragma unroll
        for (int off = 1; off < 16; off <<= 1) {
#pragma unroll
            for (int r = 0; r < 8; ++r) {
                float om = __shfl_xor(mr[r], off, 16);
                float os = __shfl_xor(sr[r], off, 16);
                float mn = fmaxf(mr[r], om);
                sr[r] = sr[r] * __expf(mr[r] - mn) + os * __expf(om - mn);
                mr[r] = mn;
            }
        }
#pragma unroll
        for (int r = 0; r < 8; ++r) dA[trr][r] = __expf(dg[r] - mr[r]) / sr[r];
    }
    __syncthreads();
    float* diagA = (float*)ks;        // K no longer needed
    for (int trr = 0; trr < 2; ++trr) {
        int m0 = wid * 32 + trr * 16;
#pragma unroll
        for (int r = 0; r < 8; ++r)
            if (r15 == h * 8 + r) diagA[m0 + h * 8 + r] = dA[trr][r];
    }
    __syncthreads();

    // ---- V projection (recompute), x2 = x + relu(diagA*V) into qs ----
    for (int trr = 0; trr < 2; ++trr) {
        int m0 = wid * 32 + trr * 16;
        v8f av[4];
#pragma unroll
        for (int j = 0; j < 4; ++j) av[j] = zero8();
#pragma unroll
        for (int kk = 0; kk < 2; ++kk) {
            int k0 = kk * 32;
            v16bf a = load_a16(xs, 64, m0, k0, lane);
#pragma unroll
            for (int j = 0; j < 4; ++j)
                av[j] = wmma_bf16(a, load_bt16(Wv, 64, j * 16, k0, lane), av[j]);
        }
#pragma unroll
        for (int j = 0; j < 4; ++j) {
            int f = j * 16 + r15;
            float bvv = bv[f];
#pragma unroll
            for (int r = 0; r < 8; ++r) {
                int nr = m0 + h * 8 + r;
                float vv = av[j][r] + bvv;
                float g  = fmaxf(diagA[nr] * vv, 0.f);
                float x2 = (float)xs[nr * 64 + f] + g;
                qs[nr * 64 + f] = (__bf16)x2;
            }
        }
    }
    __syncthreads();

    // ---- x3 = x2*Wd + bd; inf-norm rows; x4 = 0.5*(x3/norm + 1); histogram ----
    for (int trr = 0; trr < 2; ++trr) {
        int m0 = wid * 32 + trr * 16;
        v8f ad[4];
#pragma unroll
        for (int j = 0; j < 4; ++j) ad[j] = zero8();
#pragma unroll
        for (int kk = 0; kk < 2; ++kk) {
            int k0 = kk * 32;
            v16bf a = load_a16(qs, 64, m0, k0, lane);
#pragma unroll
            for (int j = 0; j < 4; ++j)
                ad[j] = wmma_bf16(a, load_bt16(Wd, 64, j * 16, k0, lane), ad[j]);
        }
        float val[4][8], amax[8];
#pragma unroll
        for (int r = 0; r < 8; ++r) amax[r] = 0.f;
#pragma unroll
        for (int j = 0; j < 4; ++j) {
            float bdv = bd[j * 16 + r15];
#pragma unroll
            for (int r = 0; r < 8; ++r) {
                val[j][r] = ad[j][r] + bdv;
                amax[r] = fmaxf(amax[r], fabsf(val[j][r]));
            }
        }
#pragma unroll
        for (int off = 1; off < 16; off <<= 1)
#pragma unroll
            for (int r = 0; r < 8; ++r)
                amax[r] = fmaxf(amax[r], __shfl_xor(amax[r], off, 16));
#pragma unroll
        for (int j = 0; j < 4; ++j) {
            int f = j * 16 + r15;
#pragma unroll
            for (int r = 0; r < 8; ++r) {
                int nr = m0 + h * 8 + r;
                float xv = 0.5f * (val[j][r] / amax[r] + 1.0f);
                x4[((size_t)bt * N_ + nr) * F_ + f] = xv;
                int bin = (int)(xv * 4096.0f);
                bin = bin < 0 ? 0 : (bin > NBIN - 1 ? NBIN - 1 : bin);
                atomicAdd(&hist1[b * NBIN + bin], 1);
            }
        }
    }
}

// ---------------- kernel 4/6: select threshold bin from histogram ----------------
__global__ void mm_sel1(const int* hist1, float* lo, int* cab) {
    int b = blockIdx.x;
    if (threadIdx.x != 0) return;
    const int target = KTOP + 1;
    int cum = 0, found = 0, ca = 0;
    float lov = 0.f;
    for (int bin = NBIN - 1; bin >= 0; --bin) {
        int c = hist1[b * NBIN + bin];
        if (cum + c >= target) { lov = (float)bin / 4096.0f; ca = cum; found = 1; break; }
        cum += c;
    }
    if (!found) { lov = 0.f; ca = cum; }
    lo[b] = lov; cab[b] = ca;
}

// ---------------- kernel 5: refine within candidate bin -------------------------
__global__ void mm_hist2(const float* __restrict__ x4, const float* __restrict__ lo,
                         int* __restrict__ hist2) {
    size_t i = (size_t)blockIdx.x * blockDim.x + threadIdx.x;
    size_t stride = (size_t)gridDim.x * blockDim.x;
    for (; i < TOTAL; i += stride) {
        int b = (int)(i >> 20);       // NTF = 2^20
        float rel = (x4[i] - lo[b]) * 4096.0f;
        if (rel >= 0.f && rel < 1.f) {
            int sb = (int)(rel * 4096.0f);
            sb = sb > NBIN - 1 ? NBIN - 1 : sb;
            atomicAdd(&hist2[b * NBIN + sb], 1);
        }
    }
}

__global__ void mm_sel2(const int* hist2, const float* lo, const int* cab, float* mu) {
    int b = blockIdx.x;
    if (threadIdx.x != 0) return;
    const int target = KTOP + 1;
    int cum = cab[b];
    float m = lo[b];
    for (int sb = NBIN - 1; sb >= 0; --sb) {
        cum += hist2[b * NBIN + sb];
        if (cum >= target) { m = lo[b] + (float)sb / (4096.0f * 4096.0f); break; }
    }
    mu[b] = m;
}

// ---------------- kernel 7: sigmoid + STE mask + Wp projection ------------------
__global__ __launch_bounds__(256) void mm_final(float* __restrict__ x4m,   // in: x4, out: mask
                                                const __bf16* __restrict__ wt,
                                                const float* __restrict__ bp,
                                                const float* __restrict__ mu,
                                                float* __restrict__ pred) {
    __shared__ __align__(16) __bf16 ms[256 * 64];
    int bt = blockIdx.x;
    int b  = bt >> 6;
    float muv = mu[b];
    int tid = threadIdx.x;
    float* slab = x4m + (size_t)bt * (N_ * F_);
    for (int e = tid; e < N_ * F_; e += 256) {
        float v  = slab[e];
        float s  = 1.0f / (1.0f + __expf(muv - v));
        float mk = (s >= 0.5f) ? 1.0f : 0.0f;
        ms[e]   = (__bf16)(s * mk);
        slab[e] = mk;                 // overwrite x4 with mask (same thread/addr)
    }
    __syncthreads();
    const __bf16* Wp = wt + 11 * 4096;
    int wid = tid >> 5, lane = tid & 31, h = lane >> 4, r15 = lane & 15;
    for (int trr = 0; trr < 2; ++trr) {
        int m0 = wid * 32 + trr * 16;
        v8f ap[4];
#pragma unroll
        for (int j = 0; j < 4; ++j) ap[j] = zero8();
#pragma unroll
        for (int kk = 0; kk < 2; ++kk) {
            int k0 = kk * 32;
            v16bf a = load_a16(ms, 64, m0, k0, lane);
#pragma unroll
            for (int j = 0; j < 4; ++j)
                ap[j] = wmma_bf16(a, load_bt16(Wp, 64, j * 16, k0, lane), ap[j]);
        }
#pragma unroll
        for (int j = 0; j < 4; ++j) {
            int f = j * 16 + r15;
            float bpv = bp[f];
#pragma unroll
            for (int r = 0; r < 8; ++r) {
                int nr = m0 + h * 8 + r;
                pred[((size_t)bt * N_ + nr) * F_ + f] = ap[j][r] + bpv;
            }
        }
    }
}

extern "C" void kernel_launch(void* const* d_in, const int* in_sizes, int n_in,
                              void* d_out, int out_size, void* d_ws, size_t ws_size,
                              hipStream_t stream) {
    (void)in_sizes; (void)n_in; (void)out_size; (void)ws_size;
    const float* x   = (const float*)d_in[0];
    const float* Wc1 = (const float*)d_in[1];
    const float* bc1 = (const float*)d_in[2];
    const float* Wc2 = (const float*)d_in[3];
    const float* bc2 = (const float*)d_in[4];
    const float* Wc3 = (const float*)d_in[5];
    const float* bc3 = (const float*)d_in[6];
    const float* Wq  = (const float*)d_in[7];
    const float* bq  = (const float*)d_in[8];
    const float* Wk  = (const float*)d_in[9];
    const float* bk  = (const float*)d_in[10];
    const float* Wv  = (const float*)d_in[11];
    const float* bv  = (const float*)d_in[12];
    const float* Wd  = (const float*)d_in[13];
    const float* bd  = (const float*)d_in[14];
    const float* Wp  = (const float*)d_in[15];
    const float* bp  = (const float*)d_in[16];

    char* ws = (char*)d_ws;
    __bf16* wt   = (__bf16*)ws;                                   // 12*4096 bf16
    float*  bcv  = (float*)(ws + 98304);
    int*    h1   = (int*)(ws + 99328);                            // 32*4096
    int*    h2   = (int*)(ws + 99328 + 4 * 32 * NBIN);            // 32*4096
    float*  lo   = (float*)(ws + 99328 + 8 * 32 * NBIN);
    int*    cab  = (int*)(ws + 99328 + 8 * 32 * NBIN + 128);
    float*  mu   = (float*)(ws + 99328 + 8 * 32 * NBIN + 256);

    float*  outp = (float*)d_out;
    __bf16* x1   = (__bf16*)d_out;               // stage X1 (bf16) in pred half
    float*  x4   = outp + TOTAL;                 // stage X4 in mask half

    mm_zero<<<(2 * 32 * NBIN + 255) / 256, 256, 0, stream>>>(h1, 2 * 32 * NBIN);
    mm_prep<<<(12 * 4096 + 64 + 255) / 256, 256, 0, stream>>>(
        Wc1, Wc2, Wc3, bc1, bc2, bc3, Wq, Wk, Wv, Wd, Wp, wt, bcv);
    mm_conv<<<B_ * N_, 128, 0, stream>>>(x, wt, bcv, x1);
    mm_attn<<<B_ * T_, 256, 0, stream>>>(x1, wt, bq, bk, bv, bd, x4, h1);
    mm_sel1<<<B_, 32, 0, stream>>>(h1, lo, cab);
    mm_hist2<<<4096, 256, 0, stream>>>(x4, lo, h2);
    mm_sel2<<<B_, 32, 0, stream>>>(h2, lo, cab, mu);
    mm_final<<<B_ * T_, 256, 0, stream>>>(x4, wt, bp, mu, outp);
}
